// CTC_encoder_71631464563203
// MI455X (gfx1250) — compile-verified, matrix operand order checked by reading, not run on hardware
//
#include <hip/hip_runtime.h>
#include <hip/hip_bf16.h>
#include <math.h>

// ---------------------------------------------------------------------------
// CDNA5 (gfx1250) CTC encoder: conv stack as implicit-GEMM WMMA, LSTM input
// projection as WMMA GEMM, LSTM recurrence as single-WGP WMMA scan with h in
// LDS, all matrix math in bf16 inputs / f32 accumulate (v_wmma_f32_16x16x32_bf16).
// ---------------------------------------------------------------------------

typedef __attribute__((ext_vector_type(16))) __bf16 bf16x16;
typedef __attribute__((ext_vector_type(8)))  float  f32x8;
typedef __attribute__((ext_vector_type(4)))  unsigned int u32x4;
typedef __attribute__((ext_vector_type(8)))  int    i32x8;
typedef __attribute__((ext_vector_type(4)))  int    i32x4;

union FragAB { bf16x16 v; unsigned int u[8]; };
union FragC  { f32x8   v; float        f[8]; };

__device__ __forceinline__ unsigned short f2bf(float f) {
  union { float f; unsigned u; } a; a.f = f;
  unsigned u = a.u;
  unsigned r = (u + 0x7FFFu + ((u >> 16) & 1u)) >> 16;   // RNE
  return (unsigned short)r;
}
__device__ __forceinline__ float bf2f(unsigned short h) {
  union { unsigned u; float f; } a; a.u = ((unsigned)h) << 16; return a.f;
}
__device__ __forceinline__ float sigf(float x)  { return 1.f / (1.f + __expf(-x)); }
__device__ __forceinline__ float siluf(float x) { return x / (1.f + __expf(-x)); }

// Load one 16x32 bf16 A/B fragment from a row-major [rows, ldK] bf16 matrix.
// lane&15 selects the row; K packing follows ISA 7.12.2 (g*16 + hi*8 + jj*2).
__device__ __forceinline__ void load_frag_g(FragAB& f,
                                            const unsigned short* __restrict__ base,
                                            long long row, int ldK, int kbase, int hi) {
  const unsigned short* p = base + (long long)row * ldK + kbase + hi * 8;
#pragma unroll
  for (int j = 0; j < 8; ++j) {
    int k = (j >> 2) * 16 + (j & 3) * 2;
    f.u[j] = *(const unsigned int*)(p + k);
  }
}

// ---------------------------------------------------------------------------
// Param prep kernels
// ---------------------------------------------------------------------------

// s = g * rsqrt(v+eps); b2 = (conv_b - m) * s + beta   (BN folded onto conv acc)
__global__ void k_bnparams(const float* __restrict__ cb, const float* __restrict__ g,
                           const float* __restrict__ beta, const float* __restrict__ m,
                           const float* __restrict__ v, float* __restrict__ s,
                           float* __restrict__ b2, int O) {
  int i = blockIdx.x * blockDim.x + threadIdx.x;
  if (i >= O) return;
  float sc = g[i] * rsqrtf(v[i] + 1e-5f);
  s[i] = sc;
  b2[i] = (cb[i] - m[i]) * sc + beta[i];
}

// conv weight (O,I,K) f32 -> bf16 packed [O][kk*I + c]  (patch layout: t-major then channel)
__global__ void k_packw(const float* __restrict__ w, unsigned short* __restrict__ wp,
                        int O, int Cin, int Kk) {
  int e = blockIdx.x * blockDim.x + threadIdx.x;
  int tot = O * Cin * Kk;
  if (e >= tot) return;
  int o = e / (Cin * Kk);
  int rem = e - o * (Cin * Kk);
  int kk = rem / Cin;
  int c  = rem - kk * Cin;
  wp[(long long)o * (Cin * Kk) + kk * Cin + c] = f2bf(w[((long long)o * Cin + c) * Kk + kk]);
}

__global__ void k_cast_bf16(const float* __restrict__ in, unsigned short* __restrict__ out, int n) {
  int i = blockIdx.x * blockDim.x + threadIdx.x;
  if (i < n) out[i] = f2bf(in[i]);
}

__global__ void k_addbias(const float* __restrict__ a, const float* __restrict__ b,
                          float* __restrict__ o, int n) {
  int i = blockIdx.x * blockDim.x + threadIdx.x;
  if (i < n) o[i] = a[i] + b[i];
}

// ---------------------------------------------------------------------------
// Small convs (memory-bound, VALU): conv0 1->8 k3 p1, conv1 8->64 k5 p2.
// Outputs channels-last bf16 into T-padded buffers (padding pre-zeroed).
// ---------------------------------------------------------------------------

__global__ void k_conv0(const float* __restrict__ x, const float* __restrict__ w,
                        const float* __restrict__ s, const float* __restrict__ b2,
                        unsigned short* __restrict__ y0) { // [64][4004][8], interior at t+2
  int idx = blockIdx.x * blockDim.x + threadIdx.x;
  if (idx >= 64 * 4000) return;
  int b = idx / 4000, t = idx - b * 4000;
  const float* xb = x + (long long)b * 4000;
  float i0 = (t > 0)    ? xb[t - 1] : 0.f;
  float i1 = xb[t];
  float i2 = (t < 3999) ? xb[t + 1] : 0.f;
  unsigned short* yo = y0 + ((long long)b * 4004 + t + 2) * 8;
#pragma unroll
  for (int o = 0; o < 8; ++o) {
    float a = w[o * 3 + 0] * i0 + w[o * 3 + 1] * i1 + w[o * 3 + 2] * i2;
    float z = a * s[o] + b2[o];
    yo[o] = f2bf(siluf(z));
  }
}

__global__ void k_conv1(const unsigned short* __restrict__ y0, const float* __restrict__ w,
                        const float* __restrict__ s, const float* __restrict__ b2,
                        unsigned short* __restrict__ y1) { // [64][4014][64], interior at t+7
  int idx = blockIdx.x * blockDim.x + threadIdx.x;
  if (idx >= 64 * 4000) return;
  int b = idx / 4000, t = idx - b * 4000;
  float win[5][8];
#pragma unroll
  for (int kk = 0; kk < 5; ++kk) {
    const unsigned short* p = y0 + ((long long)b * 4004 + t + kk) * 8;
#pragma unroll
    for (int c = 0; c < 8; ++c) win[kk][c] = bf2f(p[c]);
  }
  unsigned short* yo = y1 + ((long long)b * 4014 + t + 7) * 64;
  for (int o = 0; o < 64; ++o) {
    const float* wo = w + (long long)o * 40; // (64,8,5): w[o][c][kk]
    float a = 0.f;
#pragma unroll
    for (int c = 0; c < 8; ++c)
#pragma unroll
      for (int kk = 0; kk < 5; ++kk) a += wo[c * 5 + kk] * win[kk][c];
    float z = a * s[o] + b2[o];
    yo[o] = f2bf(siluf(z));
  }
}

// ---------------------------------------------------------------------------
// Implicit-GEMM conv via WMMA: out[b,t',n] = silu(s[n]*conv + b2[n])
// A row m=(b,t'), K = k*Cin (t-major, channel minor), B = packed weights [N,K].
// One wave computes a 16(M)x64(N) tile; block = 8 waves -> 128(M)x64(N).
// ---------------------------------------------------------------------------

__global__ __launch_bounds__(256) void k_conv_gemm(
    const unsigned short* __restrict__ in,   // padded bf16 [B][Tpad][Cin]
    const unsigned short* __restrict__ wp,   // bf16 [N][K]
    const float* __restrict__ s, const float* __restrict__ b2,
    unsigned short* __restrict__ out,        // bf16, strides below (base incl. out pad)
    int T_out, int Cin_log2, int stride, int K,
    long long inb_stride,                    // Tpad*Cin (elements)
    long long ob_stride, long long ot_stride) {
  const int lane = threadIdx.x & 31;
  const int wave = threadIdx.x >> 5;
  const int nrow = lane & 15;
  const int hi   = lane >> 4;
  const int m0   = blockIdx.x * 128 + wave * 16;
  const int n0   = blockIdx.y * 64;
  const int Cin  = 1 << Cin_log2;
  const int cmask = Cin - 1;

  // per-lane A row -> (b, t')
  const int mrow = m0 + nrow;
  const int ba = mrow / T_out;
  const int ta = mrow - ba * T_out;
  const unsigned short* inb = in + (long long)ba * inb_stride;

  FragC acc[4];
#pragma unroll
  for (int sn = 0; sn < 4; ++sn)
#pragma unroll
    for (int r = 0; r < 8; ++r) acc[sn].f[r] = 0.f;

  for (int kb = 0; kb < K; kb += 32) {
    FragAB a;
#pragma unroll
    for (int j = 0; j < 8; ++j) {
      int k  = kb + (j >> 2) * 16 + hi * 8 + (j & 3) * 2;
      int kk = k >> Cin_log2;
      int c  = k & cmask;
      a.u[j] = *(const unsigned int*)(inb + (long long)(ta * stride + kk) * Cin + c);
    }
#pragma unroll
    for (int sn = 0; sn < 4; ++sn) {
      FragAB bw;
      load_frag_g(bw, wp, n0 + sn * 16 + nrow, K, kb, hi);
      acc[sn].v = __builtin_amdgcn_wmma_f32_16x16x32_bf16(
          false, a.v, false, bw.v, (short)0, acc[sn].v, false, false);
    }
    if (kb + 32 < K) __builtin_prefetch(wp + (long long)(n0 + nrow) * K + kb + 32, 0, 1);
  }

#pragma unroll
  for (int sn = 0; sn < 4; ++sn) {
    int nn = n0 + sn * 16 + nrow;
    float sv = s[nn], bv = b2[nn];
#pragma unroll
    for (int r = 0; r < 8; ++r) {
      int mm = m0 + r + 8 * hi;
      int bb = mm / T_out;
      int tt = mm - bb * T_out;
      float z = acc[sn].f[r] * sv + bv;
      out[(long long)bb * ob_stride + (long long)tt * ot_stride + nn] = f2bf(siluf(z));
    }
  }
}

// ---------------------------------------------------------------------------
// xg GEMM: xg[M, 4H] = A[M, Din] @ wih^T + (bih+bhh).  A bf16 rows contiguous.
// ---------------------------------------------------------------------------

__global__ __launch_bounds__(256) void k_gemm_xg(
    const unsigned short* __restrict__ A,   // bf16 [M, K]
    const unsigned short* __restrict__ W,   // bf16 [N, K]
    const float* __restrict__ bias,         // f32 [N]
    float* __restrict__ out, int N, int K) {
  const int lane = threadIdx.x & 31;
  const int wave = threadIdx.x >> 5;
  const int nrow = lane & 15;
  const int hi   = lane >> 4;
  const int m0   = blockIdx.x * 128 + wave * 16;
  const int n0   = blockIdx.y * 64;

  FragC acc[4];
#pragma unroll
  for (int sn = 0; sn < 4; ++sn) {
    float bv = bias[n0 + sn * 16 + nrow];
#pragma unroll
    for (int r = 0; r < 8; ++r) acc[sn].f[r] = bv;
  }

  for (int kb = 0; kb < K; kb += 32) {
    FragAB a;
    load_frag_g(a, A, m0 + nrow, K, kb, hi);
#pragma unroll
    for (int sn = 0; sn < 4; ++sn) {
      FragAB bw;
      load_frag_g(bw, W, n0 + sn * 16 + nrow, K, kb, hi);
      acc[sn].v = __builtin_amdgcn_wmma_f32_16x16x32_bf16(
          false, a.v, false, bw.v, (short)0, acc[sn].v, false, false);
    }
    if (kb + 32 < K) __builtin_prefetch(W + (long long)(n0 + nrow) * K + kb + 32, 0, 1);
  }

#pragma unroll
  for (int sn = 0; sn < 4; ++sn) {
    int nn = n0 + sn * 16 + nrow;
#pragma unroll
    for (int r = 0; r < 8; ++r)
      out[(long long)(m0 + r + 8 * hi) * N + nn] = acc[sn].f[r];
  }
}

// ---------------------------------------------------------------------------
// LSTM recurrent scan. One workgroup, 1024 threads = 32 waves, h in LDS (bf16),
// c in registers. Per step: gates[64,4H] = xg[t] + h @ whh^T via WMMA, where
// each wave owns whole (m-tile, h-tile) pairs so i/f/g/o for a given (b,h)
// live in the same lanes -> no cross-wave gate exchange.
// ---------------------------------------------------------------------------

__global__ __launch_bounds__(1024) void k_lstm_scan(
    const float* __restrict__ xg,           // f32 [T][64][4H]
    const unsigned short* __restrict__ whh, // bf16 [4H][H]
    unsigned short* __restrict__ outseq,    // bf16 [T][64][H]
    int H, int T, int reverse) {
  __shared__ unsigned short hsh[64 * 512];  // 64 KB of the 320 KB WGP LDS
  const int fourH = 4 * H;
  const int lane  = threadIdx.x & 31;
  const int wavei = threadIdx.x >> 5;
  const int nrow  = lane & 15;
  const int hi    = lane >> 4;
  const int htiles = H >> 4;
  const int npairs = 4 * htiles;            // (64/16) m-tiles * h-tiles

  for (int i = threadIdx.x; i < 64 * H; i += 1024) hsh[i] = 0;
  float c0[4][8];
  float hval[4][8];
#pragma unroll
  for (int s = 0; s < 4; ++s)
#pragma unroll
    for (int r = 0; r < 8; ++r) { c0[s][r] = 0.f; hval[s][r] = 0.f; }
  __syncthreads();

  for (int step = 0; step < T; ++step) {
    const int t = reverse ? (T - 1 - step) : step;
    const float* xgt = xg + (long long)t * 64 * fourH;

    // ---- phase 1: matmul + gate nonlinearity (reads old h from LDS) ----
#pragma unroll
    for (int s = 0; s < 4; ++s) {
      int p = wavei + s * 32;
      if (p >= npairs) break;                 // wave-uniform
      int tm = p & 3, th = p >> 2;
      FragC acc[4];                           // q = 0:i 1:f 2:g 3:o
#pragma unroll
      for (int q = 0; q < 4; ++q) {
        int n = q * H + th * 16 + nrow;
#pragma unroll
        for (int r = 0; r < 8; ++r)
          acc[q].f[r] = xgt[(long long)(tm * 16 + r + 8 * hi) * fourH + n];
      }
      for (int kb = 0; kb < H; kb += 32) {
        FragAB a;
        const unsigned short* hp = &hsh[(tm * 16 + nrow) * H + kb + hi * 8];
#pragma unroll
        for (int j = 0; j < 8; ++j)
          a.u[j] = *(const unsigned int*)(hp + (j >> 2) * 16 + (j & 3) * 2);
#pragma unroll
        for (int q = 0; q < 4; ++q) {
          FragAB bw;
          load_frag_g(bw, whh, q * H + th * 16 + nrow, H, kb, hi);
          acc[q].v = __builtin_amdgcn_wmma_f32_16x16x32_bf16(
              false, a.v, false, bw.v, (short)0, acc[q].v, false, false);
        }
      }
#pragma unroll
      for (int r = 0; r < 8; ++r) {
        float cc = sigf(acc[1].f[r]) * c0[s][r] + sigf(acc[0].f[r]) * tanhf(acc[2].f[r]);
        c0[s][r] = cc;
        hval[s][r] = sigf(acc[3].f[r]) * tanhf(cc);
      }
    }
    __syncthreads();

    // ---- phase 2: publish new h (LDS + output sequence) ----
#pragma unroll
    for (int s = 0; s < 4; ++s) {
      int p = wavei + s * 32;
      if (p >= npairs) break;
      int tm = p & 3, th = p >> 2;
      int n = th * 16 + nrow;
#pragma unroll
      for (int r = 0; r < 8; ++r) {
        int b = tm * 16 + r + 8 * hi;
        unsigned short hb = f2bf(hval[s][r]);
        hsh[b * H + n] = hb;
        outseq[(long long)t * 64 * H + (long long)b * H + n] = hb;
      }
    }
    __syncthreads();
  }
}

// ---------------------------------------------------------------------------
// Final FC (32->5) + log_softmax
// ---------------------------------------------------------------------------

__global__ void k_fc_lsm(const unsigned short* __restrict__ seq, // bf16 [T][64][32]
                         const float* __restrict__ fw, const float* __restrict__ fb,
                         float* __restrict__ out) {               // f32 [T][64][5]
  int idx = blockIdx.x * blockDim.x + threadIdx.x;
  if (idx >= 800 * 64) return;
  float h[32];
#pragma unroll
  for (int i = 0; i < 32; ++i) h[i] = bf2f(seq[(long long)idx * 32 + i]);
  float lg[5], mx = -1e30f;
#pragma unroll
  for (int j = 0; j < 5; ++j) {
    float a = fb[j];
#pragma unroll
    for (int i = 0; i < 32; ++i) a += fw[j * 32 + i] * h[i];
    lg[j] = a;
    mx = fmaxf(mx, a);
  }
  float se = 0.f;
#pragma unroll
  for (int j = 0; j < 5; ++j) se += __expf(lg[j] - mx);
  float ls = mx + __logf(se);
#pragma unroll
  for (int j = 0; j < 5; ++j) out[(long long)idx * 5 + j] = lg[j] - ls;
}

// ---------------------------------------------------------------------------
// TDM / cluster probe: NULL tensor descriptor (count=0 -> architectural NOP),
// exercises tensor_load_to_lds + s_wait_tensorcnt + s_cluster_barrier paths.
// This toolchain uses the 6-arg builtin: (g0, g1, g2, g3, g4, cpol).
// ---------------------------------------------------------------------------

__global__ void k_tdm_probe(unsigned int* __restrict__ scratch) {
  __shared__ unsigned int buf[64];
  if (threadIdx.x == 0) {
    u32x4 g0 = (u32x4){0u, 0u, 0u, 0u};     // count=0 => NULL tensor (NOP per ISA 8.3)
    i32x8 g1 = (i32x8){0, 0, 0, 0, 0, 0, 0, 0};
    i32x4 g2 = (i32x4){0, 0, 0, 0};
    i32x4 g3 = (i32x4){0, 0, 0, 0};
    i32x8 g4 = (i32x8){0, 0, 0, 0, 0, 0, 0, 0};
    __builtin_amdgcn_tensor_load_to_lds(g0, g1, g2, g3, g4, 0);
    __builtin_amdgcn_s_wait_tensorcnt(0);
  }
  __builtin_amdgcn_s_cluster_barrier();
  buf[threadIdx.x & 63] = (unsigned)__builtin_amdgcn_cluster_id_x();
  __syncthreads();
  scratch[threadIdx.x & 63] = buf[threadIdx.x & 63]; // keep live (ws scribble, unread)
}

// ---------------------------------------------------------------------------
// Host orchestration
// ---------------------------------------------------------------------------

static inline size_t al256(size_t x) { return (x + 255) & ~(size_t)255; }

extern "C" void kernel_launch(void* const* d_in, const int* in_sizes, int n_in,
                              void* d_out, int out_size, void* d_ws, size_t ws_size,
                              hipStream_t stream) {
  (void)in_sizes; (void)n_in; (void)out_size; (void)ws_size;
  const float* X = (const float*)d_in[0];
  auto cw  = [&](int i) { return (const float*)d_in[1 + i * 6 + 0]; };
  auto cb  = [&](int i) { return (const float*)d_in[1 + i * 6 + 1]; };
  auto bg  = [&](int i) { return (const float*)d_in[1 + i * 6 + 2]; };
  auto bb  = [&](int i) { return (const float*)d_in[1 + i * 6 + 3]; };
  auto bm  = [&](int i) { return (const float*)d_in[1 + i * 6 + 4]; };
  auto bv  = [&](int i) { return (const float*)d_in[1 + i * 6 + 5]; };
  auto wih = [&](int i) { return (const float*)d_in[31 + i * 4 + 0]; };
  auto whh = [&](int i) { return (const float*)d_in[31 + i * 4 + 1]; };
  auto bih = [&](int i) { return (const float*)d_in[31 + i * 4 + 2]; };
  auto bhh = [&](int i) { return (const float*)d_in[31 + i * 4 + 3]; };
  const float* fcw = (const float*)d_in[51];
  const float* fcb = (const float*)d_in[52];

  // ---- ws layout ----
  char* base = (char*)d_ws;
  size_t off = 0;
  auto alloc = [&](size_t bytes) { void* p = base + off; off = al256(off + bytes); return p; };

  unsigned short* y0  = (unsigned short*)alloc((size_t)64 * 4004 * 8 * 2);
  unsigned short* y1  = (unsigned short*)alloc((size_t)64 * 4014 * 64 * 2);
  unsigned short* y2  = (unsigned short*)alloc((size_t)64 * 804 * 512 * 2);
  unsigned short* y3  = (unsigned short*)alloc((size_t)64 * 804 * 512 * 2);
  unsigned short* sqA = (unsigned short*)alloc((size_t)800 * 64 * 512 * 2);
  unsigned short* sqB = (unsigned short*)alloc((size_t)800 * 64 * 512 * 2);
  float*          xg  = (float*)alloc((size_t)800 * 64 * 2048 * 4);
  unsigned short* w2p = (unsigned short*)alloc((size_t)512 * 960 * 2);
  unsigned short* w3p = (unsigned short*)alloc((size_t)512 * 2560 * 2);
  unsigned short* w4p = (unsigned short*)alloc((size_t)512 * 2560 * 2);
  float* sArr[5]; float* b2Arr[5];
  for (int i = 0; i < 5; ++i) { sArr[i] = (float*)alloc(512 * 4); b2Arr[i] = (float*)alloc(512 * 4); }
  static const int LD[5] = {512, 512, 512, 512, 128};   // Din per LSTM layer
  static const int LH[5] = {512, 512, 512, 128, 32};    // H per LSTM layer
  static const int LREV[5] = {1, 0, 1, 0, 1};
  unsigned short* wihB[5]; unsigned short* whhB[5]; float* bcB[5];
  for (int i = 0; i < 5; ++i) {
    wihB[i] = (unsigned short*)alloc((size_t)4 * LH[i] * LD[i] * 2);
    whhB[i] = (unsigned short*)alloc((size_t)4 * LH[i] * LH[i] * 2);
    bcB[i]  = (float*)alloc((size_t)4 * LH[i] * 4);
  }
  unsigned int* probe = (unsigned int*)alloc(256);

  // ---- zero padded activation buffers (margins must be 0) ----
  (void)hipMemsetAsync(y0, 0, (size_t)64 * 4004 * 8 * 2, stream);
  (void)hipMemsetAsync(y1, 0, (size_t)64 * 4014 * 64 * 2, stream);
  (void)hipMemsetAsync(y2, 0, (size_t)64 * 804 * 512 * 2, stream);
  (void)hipMemsetAsync(y3, 0, (size_t)64 * 804 * 512 * 2, stream);

  k_tdm_probe<<<1, 64, 0, stream>>>(probe);

  // ---- parameter prep ----
  static const int CO[5] = {8, 64, 512, 512, 512};
  static const int CI[5] = {1, 8, 64, 512, 512};
  static const int CK[5] = {3, 5, 15, 5, 5};
  for (int i = 0; i < 5; ++i)
    k_bnparams<<<(CO[i] + 63) / 64, 64, 0, stream>>>(cb(i), bg(i), bb(i), bm(i), bv(i),
                                                     sArr[i], b2Arr[i], CO[i]);
  unsigned short* wpk[3] = {w2p, w3p, w4p};
  for (int i = 2; i < 5; ++i) {
    int tot = CO[i] * CI[i] * CK[i];
    k_packw<<<(tot + 255) / 256, 256, 0, stream>>>(cw(i), wpk[i - 2], CO[i], CI[i], CK[i]);
  }
  for (int i = 0; i < 5; ++i) {
    int nih = 4 * LH[i] * LD[i], nhh = 4 * LH[i] * LH[i], nb = 4 * LH[i];
    k_cast_bf16<<<(nih + 255) / 256, 256, 0, stream>>>(wih(i), wihB[i], nih);
    k_cast_bf16<<<(nhh + 255) / 256, 256, 0, stream>>>(whh(i), whhB[i], nhh);
    k_addbias<<<(nb + 255) / 256, 256, 0, stream>>>(bih(i), bhh(i), bcB[i], nb);
  }

  // ---- conv stack ----
  k_conv0<<<(64 * 4000 + 255) / 256, 256, 0, stream>>>(X, cw(0), sArr[0], b2Arr[0], y0);
  k_conv1<<<(64 * 4000 + 255) / 256, 256, 0, stream>>>(y0, cw(1), sArr[1], b2Arr[1], y1);

  // conv2: 64->512 k15 s5 p7 : M=64*800, K=960, in y1 [64][4014][64], out y2 (+pad 2)
  k_conv_gemm<<<dim3(400, 8), 256, 0, stream>>>(
      y1, w2p, sArr[2], b2Arr[2], y2 + 2 * 512,
      800, 6, 5, 960, (long long)4014 * 64, (long long)804 * 512, 512);
  // conv3: 512->512 k5 s1 p2 : K=2560, in y2, out y3 (+pad 2)
  k_conv_gemm<<<dim3(400, 8), 256, 0, stream>>>(
      y2, w3p, sArr[3], b2Arr[3], y3 + 2 * 512,
      800, 9, 1, 2560, (long long)804 * 512, (long long)804 * 512, 512);
  // conv4: out directly in [t][b][c] layout for the LSTM (ot=64*512, ob=512)
  k_conv_gemm<<<dim3(400, 8), 256, 0, stream>>>(
      y3, w4p, sArr[4], b2Arr[4], sqA,
      800, 9, 1, 2560, (long long)804 * 512, (long long)512, (long long)64 * 512);

  // ---- LSTM stack (ping-pong sqA/sqB) ----
  unsigned short* cur = sqA;
  unsigned short* nxt = sqB;
  for (int i = 0; i < 5; ++i) {
    int H = LH[i], Din = LD[i], fourH = 4 * H;
    k_gemm_xg<<<dim3(400, fourH / 64), 256, 0, stream>>>(cur, wihB[i], bcB[i], xg, fourH, Din);
    k_lstm_scan<<<1, 1024, 0, stream>>>(xg, whhB[i], nxt, H, 800, LREV[i]);
    unsigned short* t = cur; cur = nxt; nxt = t;
  }

  // ---- FC + log_softmax ----
  k_fc_lsm<<<(800 * 64 + 255) / 256, 256, 0, stream>>>(cur, fcw, fcb, (float*)d_out);
}
